// STGCNLayer_5171140624833
// MI455X (gfx1250) — matrix operand order repaired and checked by reading, not compiled, not run on hardware
//
#include <hip/hip_runtime.h>
#include <hip/hip_bf16.h>
#include <stdint.h>

// ---------------------------------------------------------------------------
// ST-GCN layer for MI455X (gfx1250, wave32, WMMA).
//   K1: fused Conv1d(k=3,pad=1) + InstanceNorm1d + ReLU + (h @ gcn_w^T)
//       via v_wmma_f32_16x16x32_f16, producing hw in [T][N][32] layout.
//   K2..K4: degree init / edge-degree atomics / dis = rsqrt(deg)
//   K5: out = gcn_b + dis[n]^2 * hw   (self-loop term, fully coalesced)
//   K6: per-edge wave32 gather + global_atomic_add_f32 scatter (dominant cost)
// ---------------------------------------------------------------------------

typedef __attribute__((ext_vector_type(16))) _Float16 v16h;
typedef __attribute__((ext_vector_type(8)))  _Float16 v8h;
typedef __attribute__((ext_vector_type(8)))  float    v8f;

#define TT 12          // time steps
#define CC 32          // channels (in == out)
#define NODES_PER_BLK 8
#define ROWS (NODES_PER_BLK * TT)      // 96 rows of the (n,t) GEMM
#define KCONV 96                        // C_in * 3
// LDS padding strides (bank-conflict free for 16-lane row strides)
#define APAD 104
#define WPAD 104
#define YPAD 33
#define A2PAD 40
#define GPAD 40

static __device__ __forceinline__ v16h cat16(v8h lo, v8h hi) {
    return __builtin_shufflevector(lo, hi, 0,1,2,3,4,5,6,7,8,9,10,11,12,13,14,15);
}

// ---------------------------------------------------------------------------
// K1: fused temporal conv + instance-norm + relu + GCN weight GEMM
// ---------------------------------------------------------------------------
__global__ __launch_bounds__(128)
void k_fused_wmma(const float* __restrict__ x,       // [T][N][32]
                  const float* __restrict__ conv_w,  // [32][32][3]
                  const float* __restrict__ conv_b,  // [32]
                  const float* __restrict__ gamma,   // [32]
                  const float* __restrict__ beta,    // [32]
                  const float* __restrict__ gcn_w,   // [32][32]
                  float* __restrict__ hwT,           // out: [T][N][32]
                  int N)
{
    __shared__ _Float16 sA [ROWS][APAD];   // im2col A, f16   (~20 KB)
    __shared__ _Float16 sWc[CC][WPAD];     // conv W^T [cout][k] (~6.7 KB)
    __shared__ float    sY [ROWS][YPAD];   // conv output f32 (~12.7 KB)
    __shared__ _Float16 sA2[ROWS][A2PAD];  // relu(norm) f16  (~7.7 KB)
    __shared__ _Float16 sWg[CC][GPAD];     // gcn W [d][c]    (~2.6 KB)
    __shared__ float    sSa[NODES_PER_BLK][CC];  // gamma*rstd
    __shared__ float    sSb[NODES_PER_BLK][CC];  // beta - mu*gamma*rstd
    __shared__ float    sCb[CC];

    const int tid  = threadIdx.x;
    const int wave = tid >> 5;
    const int lane = tid & 31;
    const int nb   = blockIdx.x * NODES_PER_BLK;

    // ---- load weights into LDS (f16, transposed for contiguous B-frags) ----
    for (int i = tid; i < CC * KCONV; i += 128) {      // conv_w: [co][ci][3] -> sWc[co][kk*32+ci]
        int co = i / KCONV, k = i % KCONV;
        int kk = k >> 5, ci = k & 31;
        sWc[co][k] = (_Float16)conv_w[co * KCONV + ci * 3 + kk];
    }
    for (int i = tid; i < CC * CC; i += 128) {         // gcn_w: [d][c] -> sWg[d][c]
        sWg[i >> 5][i & 31] = (_Float16)gcn_w[i];
    }
    if (tid < CC) sCb[tid] = conv_b[tid];

    // ---- zero A (covers zero-padded conv borders), then im2col scatter ----
    {
        uint32_t* pa = (uint32_t*)&sA[0][0];
        for (int i = tid; i < (ROWS * APAD) / 2; i += 128) pa[i] = 0u;
    }
    __syncthreads();

    for (int i = tid; i < NODES_PER_BLK * TT * CC; i += 128) {
        int c  = i & 31;
        int t  = (i >> 5) % TT;
        int ln = i / (TT * CC);
        int n  = nb + ln;
        float v = (n < N) ? x[(size_t)t * N * CC + (size_t)n * CC + c] : 0.0f;
        _Float16 h = (_Float16)v;
        int rb = ln * TT;
        if (t + 1 < TT) sA[rb + t + 1][c]      = h;   // tap k=0 (x_{t'-1})
        sA[rb + t][32 + c]                      = h;   // tap k=1 (x_{t'})
        if (t >= 1)     sA[rb + t - 1][64 + c] = h;   // tap k=2 (x_{t'+1})
    }
    __syncthreads();

    const int mrow = lane & 15;    // row within 16-row tile
    const int kh   = lane >> 4;    // K-half select (ISA 7.12.2 A/B layouts)

    // ---- conv GEMM: 6 M-tiles x 2 N-tiles, K=96 in 3 WMMA steps ----
    for (int q = wave; q < 12; q += 4) {
        int mt = q >> 1, nt = q & 1;
        int m   = mt * 16 + mrow;
        int n16 = nt * 16 + mrow;
        v8f acc = {};
#pragma unroll
        for (int ks = 0; ks < 3; ++ks) {
            int kb = ks * 32;
            const _Float16* pa = &sA[m][kb + kh * 8];
            v16h a = cat16(*(const v8h*)pa, *(const v8h*)(pa + 16));
            const _Float16* pb = &sWc[n16][kb + kh * 8];
            v16h b = cat16(*(const v8h*)pb, *(const v8h*)(pb + 16));
            acc = __builtin_amdgcn_wmma_f32_16x16x32_f16(false, a, false, b,
                                                         (short)0, acc, false, false);
        }
        float cb = sCb[n16];
#pragma unroll
        for (int v2 = 0; v2 < 8; ++v2)
            sY[mt * 16 + kh * 8 + v2][n16] = acc[v2] + cb;   // D: row=(lane>>4)*8+v, col=lane&15
    }
    __syncthreads();

    // ---- instance-norm stats over T per (node, channel) ----
    for (int p = tid; p < NODES_PER_BLK * CC; p += 128) {
        int ln = p >> 5, c = p & 31;
        float s = 0.f, s2 = 0.f;
#pragma unroll
        for (int t = 0; t < TT; ++t) {
            float v = sY[ln * TT + t][c];
            s += v; s2 += v * v;
        }
        float mu   = s * (1.0f / TT);
        float var  = s2 * (1.0f / TT) - mu * mu;
        float rstd = rsqrtf(var + 1e-5f);
        float ga   = gamma[c] * rstd;
        sSa[ln][c] = ga;
        sSb[ln][c] = beta[c] - mu * ga;
    }
    __syncthreads();

    // ---- normalize + relu -> f16 A2 ----
    for (int i = tid; i < ROWS * CC; i += 128) {
        int c = i & 31, row = i >> 5, ln = row / TT;
        float v = sY[row][c] * sSa[ln][c] + sSb[ln][c];
        sA2[row][c] = (_Float16)fmaxf(v, 0.0f);
    }
    __syncthreads();

    // ---- GCN GEMM: hw = relu_out @ gcn_w^T, K=32 in one WMMA ----
    for (int q = wave; q < 12; q += 4) {
        int mt = q >> 1, nt = q & 1;
        int m    = mt * 16 + mrow;
        int dcol = nt * 16 + mrow;
        const _Float16* pa = &sA2[m][kh * 8];
        v16h a = cat16(*(const v8h*)pa, *(const v8h*)(pa + 16));
        const _Float16* pb = &sWg[dcol][kh * 8];
        v16h b = cat16(*(const v8h*)pb, *(const v8h*)(pb + 16));
        v8f acc = {};
        acc = __builtin_amdgcn_wmma_f32_16x16x32_f16(false, a, false, b,
                                                     (short)0, acc, false, false);
#pragma unroll
        for (int v2 = 0; v2 < 8; ++v2) {
            int row = mt * 16 + kh * 8 + v2;
            int ln = row / TT, t = row - ln * TT;
            int n = nb + ln;
            if (n < N)
                hwT[(size_t)t * N * CC + (size_t)n * CC + dcol] = acc[v2];
        }
    }
}

// ---------------------------------------------------------------------------
// Degree / normalization kernels
// ---------------------------------------------------------------------------
__global__ void k_deg_init(float* __restrict__ deg, int N) {
    int i = blockIdx.x * blockDim.x + threadIdx.x;
    if (i < N) deg[i] = 1.0f;                       // self-loop weight
}

__global__ void k_deg_scatter(const int* __restrict__ ei, const float* __restrict__ w,
                              float* __restrict__ deg, int E) {
    int e = blockIdx.x * blockDim.x + threadIdx.x;
    if (e < E) atomicAdd(&deg[ei[E + e]], w[e]);    // dst = col
}

__global__ void k_dis(const float* __restrict__ deg, float* __restrict__ dis, int N) {
    int i = blockIdx.x * blockDim.x + threadIdx.x;
    if (i < N) {
        float d = deg[i];
        dis[i] = (d > 0.0f) ? rsqrtf(d) : 0.0f;
    }
}

// ---------------------------------------------------------------------------
// K5: out = gcn_b + dis[n]^2 * hw   (same [T][N][32] layout -> coalesced)
// ---------------------------------------------------------------------------
__global__ void k_selfloop(const float* __restrict__ hwT, const float* __restrict__ dis,
                           const float* __restrict__ gcn_b, float* __restrict__ out, int N) {
    uint32_t i = blockIdx.x * blockDim.x + threadIdx.x;       // float4 index
    uint32_t total4 = (uint32_t)TT * (uint32_t)N * 8u;
    if (i >= total4) return;
    uint32_t d4 = i & 7u;
    uint32_t n  = (i >> 3) % (uint32_t)N;
    float s = dis[n]; s = s * s;
    const float4 b = ((const float4*)gcn_b)[d4];
    const float4 h = ((const float4*)hwT)[i];
    ((float4*)out)[i] = make_float4(b.x + s * h.x, b.y + s * h.y,
                                    b.z + s * h.z, b.w + s * h.w);
}

// ---------------------------------------------------------------------------
// K6: edge scatter — one wave32 per edge, lane == channel.
//     12 coalesced 128B gathers + 12 global_atomic_add_f32 per lane.
// ---------------------------------------------------------------------------
__global__ __launch_bounds__(128)
void k_edge_scatter(const int* __restrict__ ei, const float* __restrict__ w,
                    const float* __restrict__ dis, const float* __restrict__ hwT,
                    float* __restrict__ out, int N, int E) {
    int wave = threadIdx.x >> 5;
    int lane = threadIdx.x & 31;
    int e = blockIdx.x * 4 + wave;
    if (e >= E) return;
    int r = ei[e];
    int c = ei[E + e];
    float nrm = dis[r] * w[e] * dis[c];
    size_t NC = (size_t)N * CC;
    const float* src = hwT + (size_t)r * CC + lane;
    float*       dst = out + (size_t)c * CC + lane;
#pragma unroll
    for (int t = 0; t < TT; ++t) {
        if (t + 1 < TT)
            __builtin_prefetch(src + (size_t)(t + 1) * NC, 0, 0);  // global_prefetch next chunk
        float v = src[(size_t)t * NC];
        atomicAdd(&dst[(size_t)t * NC], nrm * v);
    }
}

// ---------------------------------------------------------------------------
extern "C" void kernel_launch(void* const* d_in, const int* in_sizes, int n_in,
                              void* d_out, int out_size, void* d_ws, size_t ws_size,
                              hipStream_t stream) {
    (void)n_in; (void)out_size; (void)ws_size;
    const float* x      = (const float*)d_in[0];
    const int*   ei     = (const int*)  d_in[1];
    const float* ew     = (const float*)d_in[2];
    const float* conv_w = (const float*)d_in[3];
    const float* conv_b = (const float*)d_in[4];
    const float* gamma  = (const float*)d_in[5];
    const float* beta   = (const float*)d_in[6];
    const float* gcn_w  = (const float*)d_in[7];
    const float* gcn_b  = (const float*)d_in[8];
    float* out = (float*)d_out;

    const int N = in_sizes[0] / (TT * CC);
    const int E = in_sizes[2];

    float* hwT = (float*)d_ws;                      // T*N*32 floats (76.8 MB)
    float* deg = hwT + (size_t)TT * N * CC;         // N floats
    float* dis = deg + N;                           // N floats

    // degree + normalization
    k_deg_init   <<<(N + 255) / 256, 256, 0, stream>>>(deg, N);
    k_deg_scatter<<<(E + 255) / 256, 256, 0, stream>>>(ei, ew, deg, E);
    k_dis        <<<(N + 255) / 256, 256, 0, stream>>>(deg, dis, N);

    // fused conv + IN + relu + gcn_w GEMM (WMMA)
    k_fused_wmma<<<(N + NODES_PER_BLK - 1) / NODES_PER_BLK, 128, 0, stream>>>(
        x, conv_w, conv_b, gamma, beta, gcn_w, hwT, N);

    // self-loop + bias init of the output
    {
        uint32_t total4 = (uint32_t)TT * (uint32_t)N * 8u;
        k_selfloop<<<(total4 + 255) / 256, 256, 0, stream>>>(hwT, dis, gcn_b, out, N);
    }

    // edge gather + atomic scatter (dominant, ~2.5 GB of traffic)
    k_edge_scatter<<<(E + 3) / 4, 128, 0, stream>>>(ei, ew, dis, hwT, out, N, E);
}
